// GRUModel_16569983828350
// MI455X (gfx1250) — matrix operand order repaired and checked by reading, not compiled, not run on hardware
//
#include <hip/hip_runtime.h>
#include <hip/hip_bf16.h>

// ---------------------------------------------------------------------------
// 2-layer GRU (B=128, T=1000, I=64, H=512) + FC + sigmoid for MI455X (gfx1250)
//
//  - Persistent kernel, 128 WGs x 256 threads (8 wave32 each).
//  - Per timestep, per layer: P[128 x 2048] = A[128 x K] @ Wcat[K x 2048]
//    with A = [h_prev | x_t] (K=576 layer0, K=1024 layer1), bf16 WMMA
//    (v_wmma_f32_16x16x32_bf16), f32 accumulate.
//  - Wcat columns interleaved per hidden unit u: (r, z, xn, hn) so each WG's
//    128x16 strip covers 4 hidden units with all gate pre-activations ->
//    gate nonlinearity is WG-local (LDS staging), no cross-WG exchange.
//  - Wcat slices for BOTH layers live in LDS for the whole kernel.
//  - Double-buffered A matrices -> 2 device-wide barriers per timestep.
//  - K-loop is software-pipelined (2-stage) with explicit global-AS loads.
// ---------------------------------------------------------------------------

typedef __attribute__((ext_vector_type(16))) __bf16 v16bf;
typedef __attribute__((ext_vector_type(8)))  float  v8f;
typedef __attribute__((ext_vector_type(4)))  unsigned uint4v;
typedef unsigned short ushort_t;

#define B_   128
#define T_   1000
#define I_   64
#define H_   512
#define K0_  576      // H + I   (layer 0 fused K)
#define K1_  1024     // H + H   (layer 1 fused K)
#define NCOL 2048     // 4*H output columns: (r,z,xn,hn) interleaved per unit
#define NWG  128
#define TPB  256

static __device__ __forceinline__ ushort_t f2bf(float f) {
  __bf16 h = (__bf16)f;
  ushort_t s;
  __builtin_memcpy(&s, &h, 2);
  return s;
}

// ---------------- device-wide sense-reversing barrier ----------------------
static __device__ __forceinline__ void grid_barrier(unsigned* cnt, unsigned* gen) {
  __threadfence();
  __syncthreads();
  if (threadIdx.x == 0) {
    unsigned g = __hip_atomic_load(gen, __ATOMIC_RELAXED, __HIP_MEMORY_SCOPE_AGENT);
    unsigned a = __hip_atomic_fetch_add(cnt, 1u, __ATOMIC_ACQ_REL, __HIP_MEMORY_SCOPE_AGENT);
    if (a == NWG - 1) {
      __hip_atomic_store(cnt, 0u, __ATOMIC_RELAXED, __HIP_MEMORY_SCOPE_AGENT);
      __hip_atomic_store(gen, g + 1u, __ATOMIC_RELEASE, __HIP_MEMORY_SCOPE_AGENT);
    } else {
      while (__hip_atomic_load(gen, __ATOMIC_ACQUIRE, __HIP_MEMORY_SCOPE_AGENT) == g) {
        __builtin_amdgcn_s_sleep(1);
      }
    }
  }
  __syncthreads();
  __threadfence();
}

// ------------- one 16x16 output tile of A[128xK] @ B[Kx16] -----------------
// A: row-major bf16 (raw u16 bits), stride Ktot. ldsB: column-major [16][Ktot].
// Fragment layouts used here (k = 16*hi + e for A and B alike, not the ISA table order):
//  A 16x32 bf16: lane m=l&15; per lane the needed halves are two contiguous
//    16B chunks at byte offsets hi*32 and 16+hi*32 (hi = l>>4).
//  B 32x16 bf16: lane n=l&15; element e -> K = 16*hi + e (32B contiguous).
//  D f32:        elem v -> row = 16*wave + v + 8*hi, col = l&15.
struct AQ { uint4v q0, q1; };
union  AU { AQ f; v16bf v; };
typedef const __attribute__((address_space(1))) uint4v* gq_t;

static __device__ __forceinline__ v8f gemm_tile(const ushort_t* __restrict__ A,
                                                const ushort_t* ldsB,
                                                const int Ktot, int wave, int lane) {
  v8f acc = {};
  const int hi = lane >> 4;
  // per-lane base: row (16*wave + m), pre-offset by hi*32 bytes
  const char* abyte =
      (const char*)(A + (size_t)(16 * wave + (lane & 15)) * Ktot) + hi * 32;
  const ushort_t* bcol = ldsB + (size_t)(lane & 15) * Ktot + hi * 16;

  AU a0, a1;
  v16bf b0, b1;
  // prologue: stage 0
  a0.f.q0 = *(gq_t)(abyte + 0);
  a0.f.q1 = *(gq_t)(abyte + 16);
  b0 = *(const v16bf*)(bcol + 0);

#pragma unroll
  for (int k0 = 0; k0 < Ktot; k0 += 64) {   // Ktot/32 is even for both layers
    // prefetch stage 1 (k0+32)
    a1.f.q0 = *(gq_t)(abyte + 2 * (k0 + 32) + 0);
    a1.f.q1 = *(gq_t)(abyte + 2 * (k0 + 32) + 16);
    b1 = *(const v16bf*)(bcol + (k0 + 32));
    acc = __builtin_amdgcn_wmma_f32_16x16x32_bf16(
        false, a0.v, false, b0, (short)0, acc, false, false);
    // prefetch stage 0 for next pair (k0+64)
    if (k0 + 64 < Ktot) {
      a0.f.q0 = *(gq_t)(abyte + 2 * (k0 + 64) + 0);
      a0.f.q1 = *(gq_t)(abyte + 2 * (k0 + 64) + 16);
      b0 = *(const v16bf*)(bcol + (k0 + 64));
    }
    acc = __builtin_amdgcn_wmma_f32_16x16x32_bf16(
        false, a1.v, false, b1, (short)0, acc, false, false);
  }
  return acc;
}

// ---------------------------- prep kernel ----------------------------------
// Builds bf16 Wcat0 [2048][576], Wcat1 [2048][1024] (column c contiguous in k;
// k<512 = h-part from W_hh^T, k>=512 = x-part from W_ih^T; c = 4u+comp with
// comp 0:r(x+h) 1:z(x+h) 2:xn(x only) 3:hn(h only)), fused bias vectors,
// initial A buffers, zero hidden states, zero barrier.
__global__ void gru_prep(const float* __restrict__ x,
                         const float* __restrict__ Wih0, const float* __restrict__ Whh0,
                         const float* __restrict__ bih0, const float* __restrict__ bhh0,
                         const float* __restrict__ Wih1, const float* __restrict__ Whh1,
                         const float* __restrict__ bih1, const float* __restrict__ bhh1,
                         ushort_t* Wc0, ushort_t* Wc1, float* Bias0, float* Bias1,
                         ushort_t* A0a, ushort_t* A1a, float* H0, float* H1,
                         unsigned* bar) {
  const int stride = gridDim.x * blockDim.x;
  const int gid0 = blockIdx.x * blockDim.x + threadIdx.x;

  // Wcat layer 0
  for (int idx = gid0; idx < NCOL * K0_; idx += stride) {
    int c = idx / K0_, k = idx - c * K0_;
    int u = c >> 2, comp = c & 3;
    float v = 0.f;
    if (k < H_) {
      if (comp == 0)      v = Whh0[(size_t)(0 * H_ + u) * H_ + k];
      else if (comp == 1) v = Whh0[(size_t)(1 * H_ + u) * H_ + k];
      else if (comp == 3) v = Whh0[(size_t)(2 * H_ + u) * H_ + k];
    } else {
      int kx = k - H_;
      if (comp == 0)      v = Wih0[(size_t)(0 * H_ + u) * I_ + kx];
      else if (comp == 1) v = Wih0[(size_t)(1 * H_ + u) * I_ + kx];
      else if (comp == 2) v = Wih0[(size_t)(2 * H_ + u) * I_ + kx];
    }
    Wc0[idx] = f2bf(v);
  }
  // Wcat layer 1
  for (int idx = gid0; idx < NCOL * K1_; idx += stride) {
    int c = idx / K1_, k = idx - c * K1_;
    int u = c >> 2, comp = c & 3;
    float v = 0.f;
    if (k < H_) {
      if (comp == 0)      v = Whh1[(size_t)(0 * H_ + u) * H_ + k];
      else if (comp == 1) v = Whh1[(size_t)(1 * H_ + u) * H_ + k];
      else if (comp == 3) v = Whh1[(size_t)(2 * H_ + u) * H_ + k];
    } else {
      int kx = k - H_;
      if (comp == 0)      v = Wih1[(size_t)(0 * H_ + u) * H_ + kx];
      else if (comp == 1) v = Wih1[(size_t)(1 * H_ + u) * H_ + kx];
      else if (comp == 2) v = Wih1[(size_t)(2 * H_ + u) * H_ + kx];
    }
    Wc1[idx] = f2bf(v);
  }
  // Fused biases (r,z: b_ih+b_hh; xn: b_ih[n]; hn: b_hh[n])
  for (int c = gid0; c < NCOL; c += stride) {
    int u = c >> 2, comp = c & 3;
    float v0, v1;
    if (comp == 0)      { v0 = bih0[u] + bhh0[u];             v1 = bih1[u] + bhh1[u]; }
    else if (comp == 1) { v0 = bih0[H_ + u] + bhh0[H_ + u];   v1 = bih1[H_ + u] + bhh1[H_ + u]; }
    else if (comp == 2) { v0 = bih0[2 * H_ + u];              v1 = bih1[2 * H_ + u]; }
    else                { v0 = bhh0[2 * H_ + u];              v1 = bhh1[2 * H_ + u]; }
    Bias0[c] = v0; Bias1[c] = v1;
  }
  // A0 buffer 0: [h0(-1)=0 | x_0]
  for (int idx = gid0; idx < B_ * K0_; idx += stride) {
    int b = idx / K0_, k = idx - b * K0_;
    float v = (k < H_) ? 0.f : x[((size_t)b * T_) * I_ + (k - H_)];
    A0a[idx] = f2bf(v);
  }
  // A1 buffer 0: zeros (h1(-1)=0; x-part filled at t=0)
  for (int idx = gid0; idx < B_ * K1_; idx += stride) A1a[idx] = f2bf(0.f);
  // hidden states
  for (int idx = gid0; idx < B_ * H_; idx += stride) { H0[idx] = 0.f; H1[idx] = 0.f; }
  if (gid0 < 8) bar[gid0] = 0u;
}

// -------------------------- persistent kernel ------------------------------
__global__ void __launch_bounds__(TPB, 1)
gru_persistent(const float* __restrict__ x,
               const float* __restrict__ Wfc, const float* __restrict__ bfc,
               const ushort_t* __restrict__ Wc0, const ushort_t* __restrict__ Wc1,
               const float* __restrict__ Bias0, const float* __restrict__ Bias1,
               ushort_t* A0a, ushort_t* A0b, ushort_t* A1a, ushort_t* A1b,
               float* H0, float* H1, unsigned* bar, float* out) {
  __shared__ ushort_t ldsB0[16 * K0_];   // 18 KB: this WG's Wcat0 slice
  __shared__ ushort_t ldsB1[16 * K1_];   // 32 KB: this WG's Wcat1 slice
  __shared__ float    P[B_ * 16];        //  8 KB: 128x16 pre-activation strip
  __shared__ float    bias0s[16];
  __shared__ float    bias1s[16];

  const int g = blockIdx.x;
  const int tid = threadIdx.x;
  const int wave = tid >> 5;
  const int lane = tid & 31;
  const int hi = lane >> 4;
  const int cL = lane & 15;

  // Persistent weight slices -> LDS (contiguous: global layout is col-major)
  for (int i = tid; i < 16 * K0_; i += TPB) ldsB0[i] = Wc0[(size_t)g * 16 * K0_ + i];
  for (int i = tid; i < 16 * K1_; i += TPB) ldsB1[i] = Wc1[(size_t)g * 16 * K1_ + i];
  if (tid < 16) { bias0s[tid] = Bias0[16 * g + tid]; bias1s[tid] = Bias1[16 * g + tid]; }
  __syncthreads();

  ushort_t* A0buf[2] = {A0a, A0b};
  ushort_t* A1buf[2] = {A1a, A1b};

#pragma unroll 1
  for (int t = 0; t < T_; ++t) {
    ushort_t* A0r = A0buf[t & 1];
    ushort_t* A0w = A0buf[(t + 1) & 1];
    ushort_t* A1r = A1buf[t & 1];
    ushort_t* A1w = A1buf[(t + 1) & 1];

    // ---- layer 0: P = [h0|x_t] @ Wcat0 strip ----
    {
      v8f acc = gemm_tile(A0r, ldsB0, K0_, wave, lane);
#pragma unroll
      for (int v = 0; v < 8; ++v)
        P[(16 * wave + v + hi * 8) * 16 + cL] = acc[v];
    }
    __syncthreads();
    // ---- layer 0 gate update (4 hidden units owned by this WG) ----
    for (int q = tid; q < 512; q += TPB) {
      int b = q >> 2, j = q & 3;
      float pr  = P[b * 16 + 4 * j + 0] + bias0s[4 * j + 0];
      float pz  = P[b * 16 + 4 * j + 1] + bias0s[4 * j + 1];
      float pxn = P[b * 16 + 4 * j + 2] + bias0s[4 * j + 2];
      float phn = P[b * 16 + 4 * j + 3] + bias0s[4 * j + 3];
      float r = 1.f / (1.f + __expf(-pr));
      float z = 1.f / (1.f + __expf(-pz));
      float n = tanhf(pxn + r * phn);
      int u = 4 * g + j;
      float hprev = H0[b * H_ + u];
      float hnew = (1.f - z) * n + z * hprev;
      H0[b * H_ + u] = hnew;
      ushort_t hb = f2bf(hnew);
      A0w[(size_t)b * K0_ + u] = hb;        // h-part for next timestep, layer 0
      A1r[(size_t)b * K1_ + H_ + u] = hb;   // x-part for this timestep, layer 1
    }
    // prefetch x_{t+1} into the write buffer
    if (t + 1 < T_) {
      int gt = g * TPB + tid;
      if (gt < B_ * I_) {
        int b = gt >> 6, c = gt & 63;
        A0w[(size_t)b * K0_ + H_ + c] = f2bf(x[((size_t)b * T_ + (t + 1)) * I_ + c]);
      }
    }
    grid_barrier(bar, bar + 1);

    // ---- layer 1: P = [h1|h0_t] @ Wcat1 strip ----
    {
      v8f acc = gemm_tile(A1r, ldsB1, K1_, wave, lane);
#pragma unroll
      for (int v = 0; v < 8; ++v)
        P[(16 * wave + v + hi * 8) * 16 + cL] = acc[v];
    }
    __syncthreads();
    // ---- layer 1 gate update ----
    for (int q = tid; q < 512; q += TPB) {
      int b = q >> 2, j = q & 3;
      float pr  = P[b * 16 + 4 * j + 0] + bias1s[4 * j + 0];
      float pz  = P[b * 16 + 4 * j + 1] + bias1s[4 * j + 1];
      float pxn = P[b * 16 + 4 * j + 2] + bias1s[4 * j + 2];
      float phn = P[b * 16 + 4 * j + 3] + bias1s[4 * j + 3];
      float r = 1.f / (1.f + __expf(-pr));
      float z = 1.f / (1.f + __expf(-pz));
      float n = tanhf(pxn + r * phn);
      int u = 4 * g + j;
      float hprev = H1[b * H_ + u];
      float hnew = (1.f - z) * n + z * hprev;
      H1[b * H_ + u] = hnew;
      A1w[(size_t)b * K1_ + u] = f2bf(hnew);
    }
    grid_barrier(bar, bar + 1);
  }

  // ---- FC + sigmoid head (O=1), WG 0 only ----
  if (g == 0 && tid < B_) {
    float s = bfc[0];
    const float* hrow = H1 + (size_t)tid * H_;
    for (int j = 0; j < H_; ++j) s += hrow[j] * Wfc[j];
    out[tid] = 1.f / (1.f + __expf(-s));
  }
}

// ------------------------------ launcher -----------------------------------
extern "C" void kernel_launch(void* const* d_in, const int* in_sizes, int n_in,
                              void* d_out, int out_size, void* d_ws, size_t ws_size,
                              hipStream_t stream) {
  const float* x    = (const float*)d_in[0];
  const float* Wih0 = (const float*)d_in[1];
  const float* Whh0 = (const float*)d_in[2];
  const float* bih0 = (const float*)d_in[3];
  const float* bhh0 = (const float*)d_in[4];
  const float* Wih1 = (const float*)d_in[5];
  const float* Whh1 = (const float*)d_in[6];
  const float* bih1 = (const float*)d_in[7];
  const float* bhh1 = (const float*)d_in[8];
  const float* Wfc  = (const float*)d_in[9];
  const float* bfc  = (const float*)d_in[10];
  float* out = (float*)d_out;

  char* ws = (char*)d_ws;
  size_t off = 0;
  auto take = [&](size_t bytes) -> char* {
    char* p = ws + off;
    off += (bytes + 255) & ~(size_t)255;
    return p;
  };
  ushort_t* Wc0   = (ushort_t*)take((size_t)NCOL * K0_ * 2);  // 2.25 MB
  ushort_t* Wc1   = (ushort_t*)take((size_t)NCOL * K1_ * 2);  // 4.00 MB
  float*    Bias0 = (float*)take((size_t)NCOL * 4);
  float*    Bias1 = (float*)take((size_t)NCOL * 4);
  ushort_t* A0a   = (ushort_t*)take((size_t)B_ * K0_ * 2);
  ushort_t* A0b   = (ushort_t*)take((size_t)B_ * K0_ * 2);
  ushort_t* A1a   = (ushort_t*)take((size_t)B_ * K1_ * 2);
  ushort_t* A1b   = (ushort_t*)take((size_t)B_ * K1_ * 2);
  float*    H0    = (float*)take((size_t)B_ * H_ * 4);
  float*    H1    = (float*)take((size_t)B_ * H_ * 4);
  unsigned* bar   = (unsigned*)take(256);
  (void)in_sizes; (void)n_in; (void)out_size; (void)ws_size;

  gru_prep<<<dim3(1024), dim3(256), 0, stream>>>(
      x, Wih0, Whh0, bih0, bhh0, Wih1, Whh1, bih1, bhh1,
      Wc0, Wc1, Bias0, Bias1, A0a, A1a, H0, H1, bar);

  gru_persistent<<<dim3(NWG), dim3(TPB), 0, stream>>>(
      x, Wfc, bfc, Wc0, Wc1, Bias0, Bias1,
      A0a, A0b, A1a, A1b, H0, H1, bar, out);
}
